// NodeTaskHead_FitSphere_49435073577661
// MI455X (gfx1250) — matrix operand order repaired and checked by reading, not compiled, not run on hardware
//
#include <hip/hip_runtime.h>
#include <hip/hip_bf16.h>

// ---------------------------------------------------------------------------
// Problem constants (from reference): B=2, N=512, E=768, G=128, H=64
// ---------------------------------------------------------------------------
#define BB 2
#define NN 512
#define EE 768
#define GG 128
#define HH 64

typedef __attribute__((ext_vector_type(16))) __bf16 v16bf;
typedef __attribute__((ext_vector_type(8)))  float  v8f;

union bfpack { unsigned u[8]; v16bf v; };

// Pack two f32 into one dword of bf16 (truncation) — one v_perm_b32.
__device__ __forceinline__ unsigned pack_bf16(float lo, float hi) {
    union { float f; unsigned u; } a, b; a.f = lo; b.f = hi;
#if __has_builtin(__builtin_amdgcn_perm)
    // dst[15:0] = a[31:16], dst[31:16] = b[31:16]
    return __builtin_amdgcn_perm(b.u, a.u, 0x07060302u);
#else
    return (a.u >> 16) | (b.u & 0xFFFF0000u);
#endif
}

// Hardware tanh (CDNA5 v_tanh_f32 transcendental).
__device__ __forceinline__ float fast_tanh(float x) {
#if __has_builtin(__builtin_amdgcn_tanhf)
    return __builtin_amdgcn_tanhf(x);
#else
    float r;
    // v_nop covers the TRANS-op RAW hazard (1 independent op required).
    asm("v_tanh_f32 %0, %1\n\tv_nop" : "=v"(r) : "v"(x));
    return r;
#endif
}

// gelu(x) ~= 0.5 x (1 + tanh(0.79788456 x + 0.03567741 x^3))
__device__ __forceinline__ float gelu_fast(float x) {
    const float c1 = 0.7978845608028654f;
    const float c2 = 0.0356774081363001f;
    float x2 = x * x;
    float t  = fast_tanh(x * __builtin_fmaf(c2, x2, c1));
    return 0.5f * x * (1.0f + t);
}

// gelu(x) * w, with 0.5 prefolded into w (wh = 0.5*w):  (x + x*t) * wh
__device__ __forceinline__ float gelu_mul(float x, float wh) {
    const float c1 = 0.7978845608028654f;
    const float c2 = 0.0356774081363001f;
    float x2 = x * x;
    float t  = fast_tanh(x * __builtin_fmaf(c2, x2, c1));
    return __builtin_fmaf(x, t, x) * wh;
}

// ---------------------------------------------------------------------------
// Generic WMMA bf16 GEMM:  out[M,N] = act( A[M,K] @ W[N,K]^T + bias )
// One wave per block; each block computes a 16x16 output tile.
// K % 32 == 0, M % 16 == 0, N % 16 == 0. ldw = row stride of W (elements).
// act: 0 = none, 1 = gelu.
// ---------------------------------------------------------------------------
__global__ __launch_bounds__(32)
void gemm_wmma_bf16(const float* __restrict__ A, const float* __restrict__ W,
                    const float* __restrict__ bias, float* __restrict__ out,
                    int M, int K, int N, int ldw, int act)
{
    const int tileN = blockIdx.x;
    const int tileM = blockIdx.y;
    const int lane  = threadIdx.x;          // 0..31 (wave32)
    const int hlf   = lane >> 4;            // 0 or 1
    const int l     = lane & 15;
    const int m0    = tileM * 16;
    const int n0    = tileN * 16;
    const int akb   = hlf ? 16 : 0;         // A lane K-base (contiguous layout)
    const int bkb   = hlf ? 16 : 0;         // B lane K-base (contiguous layout)

    v8f acc = {};
    const float* arow = A + (size_t)(m0 + l) * K;
    const float* wrow = W + (size_t)(n0 + l) * ldw;

    for (int k0 = 0; k0 < K; k0 += 32) {
        bfpack a, b;
        const float* ap = arow + k0 + akb;
        float4 x0 = *(const float4*)(ap +  0);
        float4 x1 = *(const float4*)(ap +  4);
        float4 x2 = *(const float4*)(ap +  8);
        float4 x3 = *(const float4*)(ap + 12);
        a.u[0] = pack_bf16(x0.x, x0.y); a.u[1] = pack_bf16(x0.z, x0.w);
        a.u[2] = pack_bf16(x1.x, x1.y); a.u[3] = pack_bf16(x1.z, x1.w);
        a.u[4] = pack_bf16(x2.x, x2.y); a.u[5] = pack_bf16(x2.z, x2.w);
        a.u[6] = pack_bf16(x3.x, x3.y); a.u[7] = pack_bf16(x3.z, x3.w);

        const float* wp = wrow + k0 + bkb;
        float4 w0 = *(const float4*)(wp +  0);
        float4 w1 = *(const float4*)(wp +  4);
        float4 w2 = *(const float4*)(wp +  8);
        float4 w3 = *(const float4*)(wp + 12);
        b.u[0] = pack_bf16(w0.x, w0.y); b.u[1] = pack_bf16(w0.z, w0.w);
        b.u[2] = pack_bf16(w1.x, w1.y); b.u[3] = pack_bf16(w1.z, w1.w);
        b.u[4] = pack_bf16(w2.x, w2.y); b.u[5] = pack_bf16(w2.z, w2.w);
        b.u[6] = pack_bf16(w3.x, w3.y); b.u[7] = pack_bf16(w3.z, w3.w);

        acc = __builtin_amdgcn_wmma_f32_16x16x32_bf16(
                  false, a.v, false, b.v, (short)0, acc, false, false);
    }

    const float bv = bias ? bias[n0 + l] : 0.0f;
#pragma unroll
    for (int v = 0; v < 8; ++v) {
        int m = m0 + hlf * 8 + v;           // C/D layout: vgpr v -> row
        float x = acc[v] + bv;
        if (act) x = gelu_fast(x);
        out[(size_t)m * N + n0 + l] = x;
    }
}

// ---------------------------------------------------------------------------
// Fused edge MLP + sphere-fit kernel.
// One workgroup (256 threads, 8 waves) per (b,i). Each wave processes 16-j
// tiles: hidden = gelu( ef@W1c^T + pre_i + term_j ); fsp = hidden.w2 + b2;
// masked sphere-fit moment accumulation; in-block 3x3 solve.
// ---------------------------------------------------------------------------
__global__ __launch_bounds__(256)
void edge_fit_kernel(const float* __restrict__ ef,          // [B,N,N,G]
                     const float* __restrict__ dpos,        // [B,N,N,3]
                     const unsigned char* __restrict__ enpm,// [B,N,N] bool
                     const unsigned char* __restrict__ dem, // [N,N] bool
                     const unsigned char* __restrict__ nfam,// [B,N] bool
                     const int* __restrict__ drop_or_add,   // scalar
                     const float* __restrict__ e_w1,        // [G, 2G]
                     const float* __restrict__ e_w2,        // [1, G]
                     const float* __restrict__ e_b2,        // [1]
                     const float* __restrict__ pre_i,       // [B*N, G]
                     const float* __restrict__ term_j,      // [B*N, G]
                     float* __restrict__ out)               // [B,N,3]
{
    __shared__ __bf16 Wc[GG][136];          // W1c (ef columns), bf16, padded
    __shared__ float  Pi[GG];
    __shared__ float  accum[11];            // v3[3], A[6], eec, enpm_cnt

    const int bi   = blockIdx.x;            // b*N + i
    const int b    = bi >> 9;
    const int i    = bi & (NN - 1);
    const int tid  = threadIdx.x;
    const int wave = tid >> 5;
    const int lane = tid & 31;

    // Stage W1c = e_w1[:, G:2G] into LDS as bf16
    for (int idx = tid; idx < GG * GG; idx += 256) {
        int n = idx >> 7, k = idx & (GG - 1);
        float w = e_w1[n * (2 * GG) + GG + k];
        unsigned p = pack_bf16(w, w);
        __builtin_memcpy(&Wc[n][k], &p, 2);
    }
    if (tid < GG) Pi[tid] = pre_i[(size_t)bi * GG + tid];
    if (tid < 11) accum[tid] = 0.0f;
    __syncthreads();

    const int hlf = lane >> 4;
    const int l   = lane & 15;
    const int akb = hlf ? 16 : 0;
    const int bkb = hlf ? 16 : 0;
    const int doa = drop_or_add[0];

    float w2h[8];                            // 0.5 * w2 fragment
#pragma unroll
    for (int nt = 0; nt < 8; ++nt) w2h[nt] = 0.5f * e_w2[nt * 16 + l];
    const float b2 = e_b2[0];

    const float* efb = ef + (size_t)bi * NN * GG;

    for (int t = wave; t < NN / 16; t += 8) {
        const int j0 = t * 16;

        // Prefetch next tile of edge_features (global_prefetch_b8)
        if (t + 8 < NN / 16) {
            const float* np = efb + (size_t)((t + 8) * 16 + l) * GG;
            __builtin_prefetch(np, 0, 1);
            __builtin_prefetch(np + 64, 0, 1);
        }

        v8f acc[8];
#pragma unroll
        for (int nt = 0; nt < 8; ++nt) acc[nt] = (v8f){};

        const float* arow = efb + (size_t)(j0 + l) * GG;
#pragma unroll
        for (int kk = 0; kk < 4; ++kk) {
            bfpack a;
            const float* ap = arow + kk * 32 + akb;
            float4 x0 = *(const float4*)(ap +  0);
            float4 x1 = *(const float4*)(ap +  4);
            float4 x2 = *(const float4*)(ap +  8);
            float4 x3 = *(const float4*)(ap + 12);
            a.u[0] = pack_bf16(x0.x, x0.y); a.u[1] = pack_bf16(x0.z, x0.w);
            a.u[2] = pack_bf16(x1.x, x1.y); a.u[3] = pack_bf16(x1.z, x1.w);
            a.u[4] = pack_bf16(x2.x, x2.y); a.u[5] = pack_bf16(x2.z, x2.w);
            a.u[6] = pack_bf16(x3.x, x3.y); a.u[7] = pack_bf16(x3.z, x3.w);

#pragma unroll
            for (int nt = 0; nt < 8; ++nt) {
                v16bf bf;
                const __bf16* wp = &Wc[nt * 16 + l][kk * 32 + bkb];
#pragma unroll
                for (int q = 0; q < 16; ++q) bf[q] = wp[q];
                acc[nt] = __builtin_amdgcn_wmma_f32_16x16x32_bf16(
                              false, a.v, false, bf, (short)0, acc[nt], false, false);
            }
        }

        // Epilogue: +pre_i +term_j, gelu, dot with w2 (0.5 folded into w2h)
        const float* tj = term_j + (size_t)(b * NN + j0 + hlf * 8) * GG + l;
        float p[8];
#pragma unroll
        for (int v = 0; v < 8; ++v) p[v] = 0.0f;
#pragma unroll
        for (int nt = 0; nt < 8; ++nt) {
            const float pin = Pi[nt * 16 + l];
            const float wh  = w2h[nt];
#pragma unroll
            for (int v = 0; v < 8; ++v) {
                float x = acc[nt][v] + pin + tj[(size_t)v * GG + nt * 16];
                p[v] += gelu_mul(x, wh);
            }
        }
        // Reduce over the 16 lanes of each half-wave
#pragma unroll
        for (int v = 0; v < 8; ++v) {
            p[v] += __shfl_xor(p[v], 1, 32);
            p[v] += __shfl_xor(p[v], 2, 32);
            p[v] += __shfl_xor(p[v], 4, 32);
            p[v] += __shfl_xor(p[v], 8, 32);
        }

        // 16 active lanes handle one j each
        if (l < 8) {
            float fsp = 0.0f;
#pragma unroll
            for (int v = 0; v < 8; ++v) if (l == v) fsp = p[v];
            fsp += b2;

            const int jj = j0 + hlf * 8 + l;
            const size_t eidx = (size_t)bi * NN + jj;
            const bool en = enpm[eidx] != 0;
            const bool dm = dem[(size_t)i * NN + jj] != 0;
            if (!en) fsp = 0.0f;

            const float dx = dpos[eidx * 3 + 0];
            const float dy = dpos[eidx * 3 + 1];
            const float dz = dpos[eidx * 3 + 2];

            float v3x, v3y, v3z, axx, axy, axz, ayy, ayz, azz, ec;
            if (doa) {
                const float f  = dm ? 0.0f : fsp;
                const float px = dx * f, py = dy * f, pz = dz * f;
                const float sq = px * px + py * py + pz * pz;
                v3x = sq * px; v3y = sq * py; v3z = sq * pz;
                axx = px * px; axy = px * py; axz = px * pz;
                ayy = py * py; ayz = py * pz; azz = pz * pz;
                ec  = (en && !dm) ? 1.0f : 0.0f;
            } else {
                const float px = dx * fsp, py = dy * fsp, pz = dz * fsp;
                const float fa = dm ? fsp : 0.0f;
                const float qx = dx * fa, qy = dy * fa, qz = dz * fa;
                const float sq  = px * px + py * py + pz * pz;
                const float sqa = qx * qx + qy * qy + qz * qz;
                v3x = sq * px + sqa * qx; v3y = sq * py + sqa * qy; v3z = sq * pz + sqa * qz;
                axx = px * px + qx * qx; axy = px * py + qx * qy; axz = px * pz + qx * qz;
                ayy = py * py + qy * qy; ayz = py * pz + qy * qz; azz = pz * pz + qz * qz;
                ec  = en ? (dm ? 2.0f : 1.0f) : 0.0f;
            }
            atomicAdd(&accum[0], v3x); atomicAdd(&accum[1], v3y); atomicAdd(&accum[2], v3z);
            atomicAdd(&accum[3], axx); atomicAdd(&accum[4], axy); atomicAdd(&accum[5], axz);
            atomicAdd(&accum[6], ayy); atomicAdd(&accum[7], ayz); atomicAdd(&accum[8], azz);
            atomicAdd(&accum[9], ec);
            atomicAdd(&accum[10], en ? 1.0f : 0.0f);
        }
    }
    __syncthreads();

    // In-block 3x3 solve (Cramer) and output
    if (tid == 0) {
        const float eps = 1e-5f;
        const float eec = accum[9] + eps;
        const float r0 = accum[0] / eec, r1 = accum[1] / eec, r2 = accum[2] / eec;
        const bool nf = nfam[bi] != 0;
        float a00, a01, a02, a11, a12, a22;
        if (nf) {
            a00 = 2.0f * accum[3] / eec; a01 = 2.0f * accum[4] / eec;
            a02 = 2.0f * accum[5] / eec; a11 = 2.0f * accum[6] / eec;
            a12 = 2.0f * accum[7] / eec; a22 = 2.0f * accum[8] / eec;
        } else {
            a00 = 2.0f; a11 = 2.0f; a22 = 2.0f; a01 = 0.0f; a02 = 0.0f; a12 = 0.0f;
        }
        const float c00 = a11 * a22 - a12 * a12;
        const float c01 = a01 * a22 - a12 * a02;
        const float c02 = a01 * a12 - a11 * a02;
        const float det = a00 * c00 - a01 * c01 + a02 * c02;
        const float inv = 1.0f / det;
        float x0 = (r0 * c00 - a01 * (r1 * a22 - a12 * r2) + a02 * (r1 * a12 - a11 * r2)) * inv;
        float x1 = (a00 * (r1 * a22 - a12 * r2) - r0 * c01 + a02 * (a01 * r2 - r1 * a02)) * inv;
        float x2 = (a00 * (a11 * r2 - r1 * a12) - a01 * (a01 * r2 - r1 * a02) + r0 * c02) * inv;
        const float npmc = accum[10];
        const bool untrust = (eec / (npmc + eps)) < 0.3f;
        float s = (untrust || !nf) ? 0.0f : 2.0f;
        out[(size_t)bi * 3 + 0] = s * x0;
        out[(size_t)bi * 3 + 1] = s * x1;
        out[(size_t)bi * 3 + 2] = s * x2;
    }
}

// ---------------------------------------------------------------------------
// Launch
// ---------------------------------------------------------------------------
extern "C" void kernel_launch(void* const* d_in, const int* in_sizes, int n_in,
                              void* d_out, int out_size, void* d_ws, size_t ws_size,
                              hipStream_t stream) {
    const float* query = (const float*)d_in[0];
    const float* ef    = (const float*)d_in[1];
    const float* dpos  = (const float*)d_in[2];
    const unsigned char* enpm = (const unsigned char*)d_in[3];
    const unsigned char* dem  = (const unsigned char*)d_in[4];
    const unsigned char* nfam = (const unsigned char*)d_in[5];
    const int*   doa   = (const int*)d_in[6];
    const float* q_w1 = (const float*)d_in[7];
    const float* q_b1 = (const float*)d_in[8];
    const float* q_w2 = (const float*)d_in[9];
    const float* q_b2 = (const float*)d_in[10];
    const float* k_w1 = (const float*)d_in[11];
    const float* k_b1 = (const float*)d_in[12];
    const float* k_w2 = (const float*)d_in[13];
    const float* k_b2 = (const float*)d_in[14];
    const float* e_w1 = (const float*)d_in[15];
    const float* e_b1 = (const float*)d_in[16];
    const float* e_w2 = (const float*)d_in[17];
    const float* e_b2 = (const float*)d_in[18];

    const int M = BB * NN;                     // 1024 rows
    char* ws = (char*)d_ws;
    float* h_buf = (float*)(ws);                                // [1024,768]
    float* gq    = (float*)(ws + (size_t)M * EE * 4);           // [1024,64]
    float* gk    = gq + (size_t)M * HH;                         // [1024,64]
    float* prei  = gk + (size_t)M * HH;                         // [1024,128]
    float* termj = prei + (size_t)M * GG;                       // [1024,128]

    const dim3 wv(32);
    // gq = gelu( gelu(query@q_w1^T+q_b1) @ q_w2^T + q_b2 )
    gemm_wmma_bf16<<<dim3(EE/16, M/16), wv, 0, stream>>>(query, q_w1, q_b1, h_buf, M, EE, EE, EE, 1);
    gemm_wmma_bf16<<<dim3(HH/16, M/16), wv, 0, stream>>>(h_buf, q_w2, q_b2, gq,    M, EE, HH, EE, 1);
    // gk likewise
    gemm_wmma_bf16<<<dim3(EE/16, M/16), wv, 0, stream>>>(query, k_w1, k_b1, h_buf, M, EE, EE, EE, 1);
    gemm_wmma_bf16<<<dim3(HH/16, M/16), wv, 0, stream>>>(h_buf, k_w2, k_b2, gk,    M, EE, HH, EE, 1);
    // pre_i = gq @ e_w1[:, :64]^T + e_b1 ; term_j = gk @ e_w1[:, 64:128]^T
    gemm_wmma_bf16<<<dim3(GG/16, M/16), wv, 0, stream>>>(gq, e_w1,      e_b1,    prei,  M, HH, GG, 2*GG, 0);
    gemm_wmma_bf16<<<dim3(GG/16, M/16), wv, 0, stream>>>(gk, e_w1 + HH, nullptr, termj, M, HH, GG, 2*GG, 0);

    // Fused edge MLP + sphere fit + 3x3 solve
    edge_fit_kernel<<<dim3(BB * NN), dim3(256), 0, stream>>>(
        ef, dpos, enpm, dem, nfam, doa, e_w1, e_w2, e_b2, prei, termj, (float*)d_out);
}
